// Attention_32985348833769
// MI455X (gfx1250) — matrix operand order repaired and checked by reading, not compiled, not run on hardware
//
#include <hip/hip_runtime.h>
#include <math.h>
#include <stdint.h>

typedef __bf16 bf16_t;
typedef __attribute__((ext_vector_type(16))) __bf16 v16bf;
typedef __attribute__((ext_vector_type(8)))  __bf16 v8bf;
typedef __attribute__((ext_vector_type(4)))  __bf16 v4bf;
typedef __attribute__((ext_vector_type(8)))  float  v8f;
typedef __attribute__((ext_vector_type(4)))  float  v4f;
typedef __attribute__((ext_vector_type(4)))  unsigned int v4u;
typedef __attribute__((ext_vector_type(8)))  int v8i;
typedef __attribute__((ext_vector_type(4)))  int v4i;

#define B_ 2
#define T_ 2048
#define E_ 1024
#define H_ 16
#define D_ 64
#define M_ (B_*T_)   // 4096 rows for the big GEMMs

union FragBF { v16bf v; v8bf h[2]; };

// ---- WMMA fragment loaders (bf16, 16x16x32) -------------------------------
// A (16x32, MxK): lane L holds row M=L&15; element i -> K = kb + i,
// kb = 0 (lanes 0-15) / 16 (lanes 16-31)
__device__ __forceinline__ v16bf load_a_frag(const bf16_t* __restrict__ base,
                                             int ld, int row0, int k0, int lane) {
  const int r  = row0 + (lane & 15);
  const int kb = k0 + ((lane & 16) ? 16 : 0);
  const bf16_t* p = base + (size_t)r * ld + kb;
  FragBF u;
  u.h[0] = *(const v8bf*)(p);        // K = kb .. kb+7
  u.h[1] = *(const v8bf*)(p + 8);    // K = kb+8 .. kb+15
  return u.v;
}

// B (32x16, KxN): lane L holds column N=L&15; element i -> K = kb + i,
// kb = 0 (lanes 0-15) / 16 (lanes 16-31). Source row index = N, K contiguous.
__device__ __forceinline__ v16bf load_b_frag(const bf16_t* __restrict__ base,
                                             int ld, int n0, int k0, int lane) {
  const int n  = n0 + (lane & 15);
  const int kb = k0 + ((lane & 16) ? 16 : 0);
  const bf16_t* p = base + (size_t)n * ld + kb;
  FragBF u;
  u.h[0] = *(const v8bf*)(p);
  u.h[1] = *(const v8bf*)(p + 8);
  return u.v;
}

__device__ __forceinline__ v8f wmma_bf16(v16bf a, v16bf b, v8f c) {
  return __builtin_amdgcn_wmma_f32_16x16x32_bf16(false, a, false, b,
                                                 (short)0, c, false, false);
}

// ---- Tensor Data Mover: 2D tile -> LDS (D# per ISA 8.3/8.4) ---------------
// data_size = 2 bytes; tile packed contiguously in LDS (row stride = tile0).
// This toolchain exposes the 6-arg builtin:
//   (uint32x4 g0, int32x8 g1, int32x4 g2, int32x4 g3, int32x8, i32 cpol)
__device__ __forceinline__ void tdm_load_2d(unsigned lds_off, const void* gptr,
                                            unsigned tile0, unsigned tile1,
                                            unsigned long long stride0) {
  const unsigned long long ga = (unsigned long long)(uintptr_t)gptr;
  v4u g0;
  g0[0] = 1u;                                           // count=1, user mode
  g0[1] = lds_off;                                      // lds_addr (bytes)
  g0[2] = (unsigned)ga;                                 // global_addr[31:0]
  g0[3] = (unsigned)((ga >> 32) & 0x01FFFFFFu)          // global_addr[56:32]
        | (2u << 30);                                   // type = 2 ("image")
  const unsigned td0 = tile0, td1 = tile1;              // tensor dims >= tile: no OOB
  v8i g1;
  g1[0] = (int)(1u << 16);                              // data_size = 1 -> 2B
  g1[1] = (int)((td0 & 0xFFFFu) << 16);                 // tensor_dim0[15:0]
  g1[2] = (int)(((td0 >> 16) & 0xFFFFu) | ((td1 & 0xFFFFu) << 16));
  g1[3] = (int)(((td1 >> 16) & 0xFFFFu) | ((tile0 & 0xFFFFu) << 16)); // tile_dim0
  g1[4] = (int)(tile1 & 0xFFFFu);                       // tile_dim1; tile_dim2=0
  g1[5] = (int)(unsigned)(stride0 & 0xFFFFFFFFull);     // tensor_dim0_stride lo
  g1[6] = (int)(unsigned)((stride0 >> 32) & 0xFFFFull); // stride hi; dim1_stride=0
  g1[7] = 0;
  const v4i z4 = {};
  const v8i z8 = {};
  __builtin_amdgcn_tensor_load_to_lds(g0, g1, z4, z4, z8, 0);
}

// ---- fp32 -> bf16 conversion (memory-bound preamble) ----------------------
__global__ void __launch_bounds__(256)
cvt_kernel(const float* __restrict__ src, bf16_t* __restrict__ dst,
           int n4, float scale) {
  int i = blockIdx.x * blockDim.x + threadIdx.x;
  if (i < n4) {
    v4f f = ((const v4f*)src)[i];
    v4bf o;
    o[0] = (bf16_t)(f[0] * scale); o[1] = (bf16_t)(f[1] * scale);
    o[2] = (bf16_t)(f[2] * scale); o[3] = (bf16_t)(f[3] * scale);
    ((v4bf*)dst)[i] = o;
  }
}

// ---- Projection GEMM: Y = X[M,K] @ W[N,K]^T, 32x64 tile per wave ----------
// transposed==0: Y[((b*H+h)*T + t)*D + d]   (Q, K)
// transposed==1: Y[((b*H+h)*D + d)*T + t]   (V, K-major for attention)
__global__ void __launch_bounds__(256)
proj_gemm(const bf16_t* __restrict__ X, const bf16_t* __restrict__ W,
          bf16_t* __restrict__ Y, int transposed) {
  const int gtid = blockIdx.x * 256 + threadIdx.x;
  const int w    = gtid >> 5;
  const int lane = threadIdx.x & 31;
  const int m0 = (w >> 4) << 5;   // 128 tiles of 32 rows
  const int n0 = (w & 15) << 6;   // 16 groups of 64 columns
  v8f acc[2][4] = {};
  for (int k0 = 0; k0 < E_; k0 += 32) {
    v16bf a0 = load_a_frag(X, E_, m0,      k0, lane);
    v16bf a1 = load_a_frag(X, E_, m0 + 16, k0, lane);
#pragma unroll
    for (int j = 0; j < 4; ++j) {
      v16bf b = load_b_frag(W, E_, n0 + 16*j, k0, lane);
      acc[0][j] = wmma_bf16(a0, b, acc[0][j]);
      acc[1][j] = wmma_bf16(a1, b, acc[1][j]);
    }
  }
  const int rbase = (lane & 16) ? 8 : 0;
  const int col   = lane & 15;
#pragma unroll
  for (int i = 0; i < 2; ++i) {
#pragma unroll
    for (int j = 0; j < 4; ++j) {
      const int n = n0 + 16*j + col;
      const int h = n >> 6, d = n & 63;
#pragma unroll
      for (int v = 0; v < 8; ++v) {
        const int m = m0 + 16*i + rbase + v;
        const int b = m >> 11, t = m & (T_-1);
        size_t idx = transposed
          ? ((size_t)(b*H_ + h) * D_ + d) * T_ + t
          : ((size_t)(b*H_ + h) * T_ + t) * D_ + d;
        Y[idx] = (bf16_t)acc[i][j][v];
      }
    }
  }
}

// ---- Flash attention: 1 wave = one 16-row query tile; TDM double-buffer ---
__global__ void __launch_bounds__(64)
flash_attn(const bf16_t* __restrict__ Qp, const bf16_t* __restrict__ Kp,
           const bf16_t* __restrict__ Vt, bf16_t* __restrict__ AO) {
  // 2 waves/block: K 2x4KB + V 2x4KB double-buffered + 1KB P bounce per wave
  __shared__ __attribute__((aligned(16))) bf16_t kbuf[2][2][32*64];
  __shared__ __attribute__((aligned(16))) bf16_t vbuf[2][2][64*32];
  __shared__ __attribute__((aligned(16))) bf16_t pbuf[2][16*32];
  const int gtid = blockIdx.x * 64 + threadIdx.x;
  const int w    = gtid >> 5;
  const int lane = threadIdx.x & 31;
  const int wid  = (threadIdx.x >> 5) & 1;
  const int bh = w >> 7;            // 128 query tiles per (b,h)
  const int q0 = (w & 127) << 4;
  const bf16_t* qb = Qp + (size_t)bh * T_ * D_;
  const bf16_t* kb = Kp + (size_t)bh * T_ * D_;
  const bf16_t* vb = Vt + (size_t)bh * D_ * T_;
  bf16_t* P = pbuf[wid];

  const v16bf qa0 = load_a_frag(qb, D_, q0, 0,  lane);
  const v16bf qa1 = load_a_frag(qb, D_, q0, 32, lane);

  v8f o[4] = {};
  float mrow[8], lrow[8];
#pragma unroll
  for (int v = 0; v < 8; ++v) { mrow[v] = -__builtin_inff(); lrow[v] = 0.f; }

  const int rbase = (lane & 16) ? 8 : 0;
  const int col   = lane & 15;

  // async DMA of the kv tiles: K tile = 32 rows x 64 (contiguous 4KB, 1D),
  // V tile = 64 rows (d) x 32 (kv), row stride T, packed to ld=32 in LDS.
  auto prefetch = [&](int kv0, int stg) {
    tdm_load_2d((unsigned)(uintptr_t)&kbuf[wid][stg][0],
                kb + (size_t)kv0 * D_, 32 * 64, 1, 32 * 64);
    tdm_load_2d((unsigned)(uintptr_t)&vbuf[wid][stg][0],
                vb + kv0, 32, 64, T_);
  };

  auto compute = [&](int kv0, int stg) {
    const bf16_t* kt = &kbuf[wid][stg][0];
    const bf16_t* vt = &vbuf[wid][stg][0];
    // S = Q @ K^T : two 16x16 column tiles, contraction over d = 64
    v8f s0 = {}, s1 = {};
    s0 = wmma_bf16(qa0, load_b_frag(kt, D_, 0,  0,  lane), s0);
    s0 = wmma_bf16(qa1, load_b_frag(kt, D_, 0,  32, lane), s0);
    s1 = wmma_bf16(qa0, load_b_frag(kt, D_, 16, 0,  lane), s1);
    s1 = wmma_bf16(qa1, load_b_frag(kt, D_, 16, 32, lane), s1);

    if (kv0 + 32 > q0) {            // causal mask on the diagonal band
#pragma unroll
      for (int v = 0; v < 8; ++v) {
        const int qrow = q0 + rbase + v;
        if (kv0 + col      > qrow) s0[v] = -__builtin_inff();
        if (kv0 + 16 + col > qrow) s1[v] = -__builtin_inff();
      }
    }

    // online softmax; row stats reduce across 16 lanes of each half-wave
#pragma unroll
    for (int v = 0; v < 8; ++v) {
      float mx = fmaxf(s0[v], s1[v]);
      mx = fmaxf(mx, __shfl_xor(mx, 1));
      mx = fmaxf(mx, __shfl_xor(mx, 2));
      mx = fmaxf(mx, __shfl_xor(mx, 4));
      mx = fmaxf(mx, __shfl_xor(mx, 8));
      const float mnew = fmaxf(mrow[v], mx);
      const float sc   = __expf(mrow[v] - mnew);
      mrow[v] = mnew;
      const float p0 = __expf(s0[v] - mnew);
      const float p1 = __expf(s1[v] - mnew);
      float ps = p0 + p1;
      ps += __shfl_xor(ps, 1);
      ps += __shfl_xor(ps, 2);
      ps += __shfl_xor(ps, 4);
      ps += __shfl_xor(ps, 8);
      lrow[v] = lrow[v] * sc + ps;
      o[0][v] *= sc; o[1][v] *= sc; o[2][v] *= sc; o[3][v] *= sc;
      const int r = rbase + v;      // C-layout -> LDS (row-major 16x32)
      P[r*32 + col]      = (bf16_t)p0;
      P[r*32 + 16 + col] = (bf16_t)p1;
    }

    __builtin_amdgcn_wave_barrier();
    asm volatile("s_wait_dscnt 0x0" ::: "memory");   // P stores -> P reads

    // O += P(16x32) @ V(32x64) from the LDS V tile (ld = 32)
    const v16bf pa = load_a_frag(P, 32, 0, 0, lane);
#pragma unroll
    for (int j = 0; j < 4; ++j) {
      v16bf vv = load_b_frag(vt, 32, j*16, 0, lane);
      o[j] = wmma_bf16(pa, vv, o[j]);
    }
    __builtin_amdgcn_wave_barrier();
    asm volatile("s_wait_dscnt 0x0" ::: "memory");   // WAR before next P stores
  };

  const int nst = (q0 >> 5) + 1;    // ceil((q0+16)/32)
  prefetch(0, 0);
  for (int s = 0; s + 1 < nst; ++s) {
    prefetch((s + 1) * 32, (s + 1) & 1);             // stage s+1 in flight
    __builtin_amdgcn_s_wait_tensorcnt(2);            // stage s landed
    compute(s * 32, s & 1);
  }
  __builtin_amdgcn_s_wait_tensorcnt(0);
  compute((nst - 1) * 32, (nst - 1) & 1);

  const int b = bh >> 4, h = bh & 15;
#pragma unroll
  for (int j = 0; j < 4; ++j) {
#pragma unroll
    for (int v = 0; v < 8; ++v) {
      const int qrow = q0 + rbase + v;
      const int d = j*16 + col;
      AO[((size_t)b*T_ + qrow)*E_ + h*D_ + d] = (bf16_t)(o[j][v] / lrow[v]);
    }
  }
}

// ---- Output GEMM: Out = A[M,E] @ Wo[E,E]^T + bo, fp32, 32x64 per wave -----
__global__ void __launch_bounds__(256)
out_gemm(const bf16_t* __restrict__ A, const bf16_t* __restrict__ W,
         const float* __restrict__ bias, float* __restrict__ Out) {
  const int gtid = blockIdx.x * 256 + threadIdx.x;
  const int w    = gtid >> 5;
  const int lane = threadIdx.x & 31;
  const int m0 = (w >> 4) << 5;
  const int n0 = (w & 15) << 6;
  v8f acc[2][4] = {};
  for (int k0 = 0; k0 < E_; k0 += 32) {
    v16bf a0 = load_a_frag(A, E_, m0,      k0, lane);
    v16bf a1 = load_a_frag(A, E_, m0 + 16, k0, lane);
#pragma unroll
    for (int j = 0; j < 4; ++j) {
      v16bf b = load_b_frag(W, E_, n0 + 16*j, k0, lane);
      acc[0][j] = wmma_bf16(a0, b, acc[0][j]);
      acc[1][j] = wmma_bf16(a1, b, acc[1][j]);
    }
  }
  const int rbase = (lane & 16) ? 8 : 0;
  const int col   = lane & 15;
#pragma unroll
  for (int i = 0; i < 2; ++i) {
#pragma unroll
    for (int j = 0; j < 4; ++j) {
      const int n = n0 + 16*j + col;
      const float bn = bias[n];
#pragma unroll
      for (int v = 0; v < 8; ++v)
        Out[(size_t)(m0 + 16*i + rbase + v) * E_ + n] = acc[i][j][v] + bn;
    }
  }
}

extern "C" void kernel_launch(void* const* d_in, const int* in_sizes, int n_in,
                              void* d_out, int out_size, void* d_ws, size_t ws_size,
                              hipStream_t stream) {
  // setup_inputs order: k, q, v, Wk, Wq, Wv, Wo, bo, mask
  const float* k_in  = (const float*)d_in[0];
  const float* q_in  = (const float*)d_in[1];
  const float* v_in  = (const float*)d_in[2];
  const float* Wk_in = (const float*)d_in[3];
  const float* Wq_in = (const float*)d_in[4];
  const float* Wv_in = (const float*)d_in[5];
  const float* Wo_in = (const float*)d_in[6];
  const float* bo_in = (const float*)d_in[7];
  float* out = (float*)d_out;

  const size_t NTE = (size_t)B_ * T_ * E_;   // 4 Mi elements
  const size_t NEE = (size_t)E_ * E_;        // 1 Mi elements
  bf16_t* kc  = (bf16_t*)d_ws;               // bf16 copies of inputs
  bf16_t* qc  = kc  + NTE;
  bf16_t* vc  = qc  + NTE;
  bf16_t* wkc = vc  + NTE;
  bf16_t* wqc = wkc + NEE;
  bf16_t* wvc = wqc + NEE;
  bf16_t* woc = wvc + NEE;
  bf16_t* Qp  = woc + NEE;                   // [B,H,T,D] (scaled by 1/8)
  bf16_t* Kp  = Qp  + NTE;                   // [B,H,T,D]
  bf16_t* Vt  = Kp  + NTE;                   // [B,H,D,T]
  bf16_t* AO  = Vt  + NTE;                   // [B,T,E] attention output
  // total = 7*NTE + 4*NEE = 32 Mi bf16 = 64 MB of workspace

  auto cvt = [&](const float* s, bf16_t* d, size_t n, float scale) {
    int n4 = (int)(n / 4);
    hipLaunchKernelGGL(cvt_kernel, dim3((n4 + 255) / 256), dim3(256), 0, stream,
                       s, d, n4, scale);
  };
  cvt(k_in,  kc,  NTE, 1.f);
  cvt(q_in,  qc,  NTE, 0.125f);   // fold 1/sqrt(D) into Q (exact power of 2)
  cvt(v_in,  vc,  NTE, 1.f);
  cvt(Wk_in, wkc, NEE, 1.f);
  cvt(Wq_in, wqc, NEE, 1.f);
  cvt(Wv_in, wvc, NEE, 1.f);
  cvt(Wo_in, woc, NEE, 1.f);

  const dim3 ggemm(256), gthr(256);          // 2048 waves, 32x64 tile each
  hipLaunchKernelGGL(proj_gemm, ggemm, gthr, 0, stream, qc, wqc, Qp, 0);
  hipLaunchKernelGGL(proj_gemm, ggemm, gthr, 0, stream, kc, wkc, Kp, 0);
  hipLaunchKernelGGL(proj_gemm, ggemm, gthr, 0, stream, vc, wvc, Vt, 1);
  hipLaunchKernelGGL(flash_attn, dim3(2048), dim3(64), 0, stream, Qp, Kp, Vt, AO);
  hipLaunchKernelGGL(out_gemm,  ggemm, gthr, 0, stream, AO, woc, bo_in, out);
}
